// conv_att_16114717294840
// MI455X (gfx1250) — matrix operand order repaired and checked by reading, not compiled, not run on hardware
//
#include <hip/hip_runtime.h>

// ---------------------------------------------------------------------------
// Non-local attention block for MI455X (gfx1250, wave32, WMMA f16->f32,
// async global->LDS staging with ASYNCcnt double buffering).
// ---------------------------------------------------------------------------

typedef __attribute__((ext_vector_type(16))) _Float16 v16h;
typedef __attribute__((ext_vector_type(8)))  float    v8f;
typedef __attribute__((ext_vector_type(4)))  unsigned int u32x4;

union V16H { v16h h; u32x4 u[2]; };

#define B_   8
#define C_   512
#define CI_  256
#define T_   2048
#define EPS_ 1e-5f

__device__ __forceinline__ u32x4 ldg16(const _Float16* p) {
    return *reinterpret_cast<const u32x4*>(p);
}

__device__ __forceinline__ v8f wmma_f16(v16h a, v16h b, v8f c) {
    // D = A(16x32 f16) * B(32x16 f16) + C(16x16 f32)
    return __builtin_amdgcn_wmma_f32_16x16x32_f16(
        /*neg_a=*/false, a, /*neg_b=*/false, b,
        /*c_mod=*/(short)0, c, /*reuse_a=*/false, /*reuse_b=*/false);
}

// Async copy of 16 bytes per lane: global -> LDS, tracked by ASYNCcnt.
__device__ __forceinline__ void async_ld16(const _Float16* lds_ptr,
                                           const _Float16* gptr) {
    unsigned lds_addr = (unsigned)(uintptr_t)lds_ptr;  // low 32b = LDS offset
    asm volatile("global_load_async_to_lds_b128 %0, %1, off"
                 :: "v"(lds_addr), "v"(gptr) : "memory");
}

// ---------------------------------------------------------------------------
// Kernel 0: zero BN accumulators
// ---------------------------------------------------------------------------
__global__ void k_zero(float* __restrict__ p, int n) {
    int i = blockIdx.x * blockDim.x + threadIdx.x;
    if (i < n) p[i] = 0.f;
}

// ---------------------------------------------------------------------------
// Kernel 1: cast the four weight matrices to f16 (same layouts)
// ---------------------------------------------------------------------------
__global__ void k_castw(const float* __restrict__ Wg, const float* __restrict__ Wt,
                        const float* __restrict__ Wp, const float* __restrict__ Wz,
                        _Float16* __restrict__ Wg_h, _Float16* __restrict__ Wt_h,
                        _Float16* __restrict__ Wp_h, _Float16* __restrict__ Wz_h) {
    int i = blockIdx.x * blockDim.x + threadIdx.x;   // < CI_*C_ == C_*CI_
    Wg_h[i] = (_Float16)Wg[i];
    Wt_h[i] = (_Float16)Wt[i];
    Wp_h[i] = (_Float16)Wp[i];
    Wz_h[i] = (_Float16)Wz[i];
}

// ---------------------------------------------------------------------------
// Kernel 2: transpose + cast x_this / x_other : (B,C,T) f32 -> (B,T,C) f16
// ---------------------------------------------------------------------------
__global__ void k_castx(const float* __restrict__ xt, const float* __restrict__ xo,
                        _Float16* __restrict__ xth, _Float16* __restrict__ xoh) {
    __shared__ _Float16 ta[32][33];
    __shared__ _Float16 tb[32][33];
    int b  = blockIdx.z;
    int c0 = blockIdx.y * 32;
    int t0 = blockIdx.x * 32;
    int tx = threadIdx.x;   // 0..31
    int ty = threadIdx.y;   // 0..7
    const size_t ibase = (size_t)b * C_ * T_;
    #pragma unroll
    for (int i = 0; i < 4; ++i) {
        int c = c0 + ty + i * 8;
        size_t off = ibase + (size_t)c * T_ + t0 + tx;
        ta[ty + i * 8][tx] = (_Float16)xt[off];
        tb[ty + i * 8][tx] = (_Float16)xo[off];
    }
    __syncthreads();
    const size_t obase = (size_t)b * T_ * C_;
    #pragma unroll
    for (int i = 0; i < 4; ++i) {
        int t = t0 + ty + i * 8;
        size_t off = obase + (size_t)t * C_ + c0 + tx;
        xth[off] = ta[tx][ty + i * 8];
        xoh[off] = tb[tx][ty + i * 8];
    }
}

// ---------------------------------------------------------------------------
// Kernel 3: projections.  One wave per (b, ci-tile, t-tile) 16x16 output tile.
//   theta[b,t,ci] = Wtheta @ x_this (+btheta) -> (B,T,CI) f16
//   phi  [b,t,ci] = Wphi   @ x_other(+bphi)   -> (B,T,CI) f16
//   g    [b,ci,t] = Wg     @ x_this (+bg)     -> (B,CI,T) f16
// ---------------------------------------------------------------------------
__global__ void __launch_bounds__(128) k_proj(
        const _Float16* __restrict__ xth, const _Float16* __restrict__ xoh,
        const _Float16* __restrict__ Wg_h, const _Float16* __restrict__ Wt_h,
        const _Float16* __restrict__ Wp_h,
        const float* __restrict__ bg, const float* __restrict__ bth,
        const float* __restrict__ bph,
        _Float16* __restrict__ theta_h, _Float16* __restrict__ phi_h,
        _Float16* __restrict__ g_h) {
    int lane  = threadIdx.x & 31;
    int wid   = blockIdx.x * 4 + (threadIdx.x >> 5);
    int b     = wid >> 11;            // 16 * 128 tiles per batch
    int rem   = wid & 2047;
    int cib   = (rem >> 7) * 16;
    int tbv   = (rem & 127) * 16;
    int lmod  = lane & 15;
    int lhalf = lane >> 4;
    int aseg  = lhalf ? 16 : 0;

    const _Float16* xtb = xth + ((size_t)b * T_ + tbv + lmod) * C_ + 16 * lhalf;
    const _Float16* xob = xoh + ((size_t)b * T_ + tbv + lmod) * C_ + 16 * lhalf;
    const _Float16* wg  = Wg_h + (size_t)(cib + lmod) * C_;
    const _Float16* wt  = Wt_h + (size_t)(cib + lmod) * C_;
    const _Float16* wp  = Wp_h + (size_t)(cib + lmod) * C_;

    v8f at = {}, ap = {}, ag = {};
    #pragma unroll
    for (int k = 0; k < 16; ++k) {
        int cb = 32 * k;
        V16H bt_, bo_, wgv, wtv, wpv;
        bt_.u[0] = ldg16(xtb + cb);            bt_.u[1] = ldg16(xtb + cb + 8);
        bo_.u[0] = ldg16(xob + cb);            bo_.u[1] = ldg16(xob + cb + 8);
        wtv.u[0] = ldg16(wt + cb + aseg);      wtv.u[1] = ldg16(wt + cb + 8 + aseg);
        wpv.u[0] = ldg16(wp + cb + aseg);      wpv.u[1] = ldg16(wp + cb + 8 + aseg);
        wgv.u[0] = ldg16(wg + cb + aseg);      wgv.u[1] = ldg16(wg + cb + 8 + aseg);
        at = wmma_f16(wtv.h, bt_.h, at);
        ap = wmma_f16(wpv.h, bo_.h, ap);
        ag = wmma_f16(wgv.h, bt_.h, ag);
    }
    int t = tbv + lmod;
    #pragma unroll
    for (int r = 0; r < 8; ++r) {
        int ci = cib + r + 8 * lhalf;
        theta_h[((size_t)b * T_ + t) * CI_ + ci]   = (_Float16)(at[r] + bth[ci]);
        phi_h  [((size_t)b * T_ + t) * CI_ + ci]   = (_Float16)(ap[r] + bph[ci]);
        g_h    [((size_t)b * CI_ + ci) * T_ + t]   = (_Float16)(ag[r] + bg[ci]);
    }
}

// ---------------------------------------------------------------------------
// Kernel 4: fused flash-attention with async-LDS double-buffered staging.
// Block = 4 waves, same batch, 4 adjacent 16-row tiles; all waves share the
// theta / g j-blocks, staged via GLOBAL_LOAD_ASYNC_TO_LDS_B128 (ASYNCcnt).
//   per j (32 columns):
//     S(16x32) = phi_rows(16x256) @ theta_cols(256x32)   [16 WMMA from LDS]
//     online softmax (shfl_xor row reductions)
//     P -> f16 via per-wave LDS staging (C-layout -> A-layout repack)
//     y(16x256) += P(16x32) @ g(32x256)                  [16 WMMA from LDS]
// ---------------------------------------------------------------------------
__global__ void __launch_bounds__(128) k_flash(
        const _Float16* __restrict__ theta_h, const _Float16* __restrict__ phi_h,
        const _Float16* __restrict__ g_h, _Float16* __restrict__ y_h) {
    __shared__ _Float16 tbuf[2][32 * 256];   // theta j-block double buffer (2x16KB)
    __shared__ _Float16 gbuf[2][256 * 32];   // g     j-block double buffer (2x16KB)
    __shared__ _Float16 pstage[4 * 512];     // per-wave P staging (4KB)

    int tid   = threadIdx.x;
    int lane  = tid & 31;
    int wib   = tid >> 5;
    int wid   = blockIdx.x * 4 + wib;
    int b     = wid >> 7;             // 128 row tiles per batch (div by 4)
    int tbv   = (wid & 127) * 16;
    int lmod  = lane & 15;
    int lhalf = lane >> 4;
    int aseg  = lhalf ? 16 : 0;
    _Float16* myl = pstage + wib * 512;

    const _Float16* thb = theta_h + (size_t)b * T_ * CI_;
    const _Float16* gb  = g_h     + (size_t)b * CI_ * T_;

    // Issue one j-block stage: exactly 16 async b128 ops per thread.
    auto issue = [&](int j, int buf) {
        const _Float16* tsrc = thb + (size_t)j * 32 * CI_;   // contiguous 16KB
        #pragma unroll
        for (int i = 0; i < 8; ++i) {
            int off = (tid + 128 * i) * 8;                   // elements
            async_ld16(&tbuf[buf][off], tsrc + off);
        }
        #pragma unroll
        for (int i = 0; i < 8; ++i) {
            int row = tid * 2 + (i >> 2);                    // ci row 0..255
            int cc  = (i & 3) * 8;                           // elements in row
            async_ld16(&gbuf[buf][row * 32 + cc],
                       gb + (size_t)row * T_ + j * 32 + cc);
        }
    };

    // Preload A operands: phi rows for this tile (fixed across j loop).
    const _Float16* phb = phi_h + ((size_t)b * T_ + tbv + lmod) * CI_;
    V16H aop[8];
    #pragma unroll
    for (int k = 0; k < 8; ++k) {
        aop[k].u[0] = ldg16(phb + 32 * k + aseg);
        aop[k].u[1] = ldg16(phb + 32 * k + 8 + aseg);
    }

    v8f zero8 = {};
    v8f yacc[16];
    #pragma unroll
    for (int i = 0; i < 16; ++i) yacc[i] = zero8;
    float mrun[8], lrun[8];
    #pragma unroll
    for (int r = 0; r < 8; ++r) { mrun[r] = -1e30f; lrun[r] = 0.f; }

    issue(0, 0);

    #pragma unroll 1
    for (int j = 0; j < 64; ++j) {
        int buf = j & 1;
        if (j + 1 < 64) {
            issue(j + 1, buf ^ 1);
            // 16 newest ops outstanding allowed; the 16 for block j are done.
            asm volatile("s_wait_asynccnt 16" ::: "memory");
        } else {
            asm volatile("s_wait_asynccnt 0" ::: "memory");
        }
        __syncthreads();   // all waves' shares of block j landed in LDS

        // ---- S = phi_rows @ theta_cols (B operands from LDS) ----
        v8f s0 = zero8, s1 = zero8;
        const _Float16* tl0 = &tbuf[buf][(size_t)lmod * 256 + 16 * lhalf];
        const _Float16* tl1 = tl0 + 16 * 256;
        #pragma unroll
        for (int k = 0; k < 8; ++k) {
            V16H b0, b1;
            b0.u[0] = *reinterpret_cast<const u32x4*>(tl0 + 32 * k);
            b0.u[1] = *reinterpret_cast<const u32x4*>(tl0 + 32 * k + 8);
            b1.u[0] = *reinterpret_cast<const u32x4*>(tl1 + 32 * k);
            b1.u[1] = *reinterpret_cast<const u32x4*>(tl1 + 32 * k + 8);
            s0 = wmma_f16(aop[k].h, b0.h, s0);
            s1 = wmma_f16(aop[k].h, b1.h, s1);
        }

        // ---- online softmax ----
        float scale[8];
        #pragma unroll
        for (int r = 0; r < 8; ++r) {
            float v = fmaxf(s0[r], s1[r]);
            v = fmaxf(v, __shfl_xor(v, 1, 32));
            v = fmaxf(v, __shfl_xor(v, 2, 32));
            v = fmaxf(v, __shfl_xor(v, 4, 32));
            v = fmaxf(v, __shfl_xor(v, 8, 32));
            float mnew = fmaxf(mrun[r], v);
            float sc   = __expf(mrun[r] - mnew);
            float p0   = __expf(s0[r] - mnew);
            float p1   = __expf(s1[r] - mnew);
            float rs   = p0 + p1;
            rs += __shfl_xor(rs, 1, 32);
            rs += __shfl_xor(rs, 2, 32);
            rs += __shfl_xor(rs, 4, 32);
            rs += __shfl_xor(rs, 8, 32);
            lrun[r]  = lrun[r] * sc + rs;
            mrun[r]  = mnew;
            scale[r] = sc;
            int row = r + 8 * lhalf;
            myl[row * 32 + lmod]      = (_Float16)p0;
            myl[row * 32 + 16 + lmod] = (_Float16)p1;
        }
        #pragma unroll
        for (int nt = 0; nt < 16; ++nt) {
            #pragma unroll
            for (int r = 0; r < 8; ++r) yacc[nt][r] *= scale[r];
        }
        asm volatile("s_wait_dscnt 0" ::: "memory");   // wave-private LDS RAW
        V16H pa;
        {
            const _Float16* pr = myl + lmod * 32 + aseg;
            pa.u[0] = *reinterpret_cast<const u32x4*>(pr);
            pa.u[1] = *reinterpret_cast<const u32x4*>(pr + 8);
        }

        // ---- y += P @ g (B operands from LDS) ----
        const _Float16* gl = &gbuf[buf][(size_t)lmod * 32 + 16 * lhalf];
        #pragma unroll
        for (int nt = 0; nt < 16; ++nt) {
            V16H bgv;
            bgv.u[0] = *reinterpret_cast<const u32x4*>(gl + nt * 512);
            bgv.u[1] = *reinterpret_cast<const u32x4*>(gl + nt * 512 + 8);
            yacc[nt] = wmma_f16(pa.h, bgv.h, yacc[nt]);
        }
        __syncthreads();   // readers done before next issue overwrites buf^1
    }

    // Normalize by row sums and store y (B,T,CI) f16.
    float linv[8];
    #pragma unroll
    for (int r = 0; r < 8; ++r) linv[r] = 1.0f / lrun[r];
    #pragma unroll
    for (int nt = 0; nt < 16; ++nt) {
        #pragma unroll
        for (int r = 0; r < 8; ++r) {
            int row = r + 8 * lhalf;
            y_h[((size_t)b * T_ + tbv + row) * CI_ + nt * 16 + lmod] =
                (_Float16)(yacc[nt][r] * linv[r]);
        }
    }
}

// ---------------------------------------------------------------------------
// Kernel 5: w_y = Wz(C,CI) @ y^T + bz -> (B,C,T) f32, plus per-channel
// sum / sum-of-squares via shfl reduction + global atomics (BN stats).
// ---------------------------------------------------------------------------
__global__ void __launch_bounds__(128) k_wz(
        const _Float16* __restrict__ y_h, const _Float16* __restrict__ Wz_h,
        const float* __restrict__ bz, float* __restrict__ wy,
        float* __restrict__ chs, float* __restrict__ chq) {
    int lane  = threadIdx.x & 31;
    int wid   = blockIdx.x * 4 + (threadIdx.x >> 5);
    int b     = wid >> 12;            // 32 * 128 tiles per batch
    int rem   = wid & 4095;
    int cb    = (rem >> 7) * 16;
    int tbv   = (rem & 127) * 16;
    int lmod  = lane & 15;
    int lhalf = lane >> 4;
    int aseg  = lhalf ? 16 : 0;

    const _Float16* wzp = Wz_h + (size_t)(cb + lmod) * CI_;
    const _Float16* yp  = y_h + ((size_t)b * T_ + tbv + lmod) * CI_ + 16 * lhalf;
    v8f acc = {};
    #pragma unroll
    for (int k = 0; k < 8; ++k) {
        V16H a, bv;
        a.u[0]  = ldg16(wzp + 32 * k + aseg);  a.u[1]  = ldg16(wzp + 32 * k + 8 + aseg);
        bv.u[0] = ldg16(yp + 32 * k);          bv.u[1] = ldg16(yp + 32 * k + 8);
        acc = wmma_f16(a.h, bv.h, acc);
    }
    int t = tbv + lmod;
    #pragma unroll
    for (int r = 0; r < 8; ++r) {
        int c = cb + r + 8 * lhalf;
        float v = acc[r] + bz[c];
        wy[((size_t)b * C_ + c) * T_ + t] = v;
        float s = v, q = v * v;
        s += __shfl_xor(s, 1, 32);  q += __shfl_xor(q, 1, 32);
        s += __shfl_xor(s, 2, 32);  q += __shfl_xor(q, 2, 32);
        s += __shfl_xor(s, 4, 32);  q += __shfl_xor(q, 4, 32);
        s += __shfl_xor(s, 8, 32);  q += __shfl_xor(q, 8, 32);
        if (lmod == 0) {
            atomicAdd(&chs[c], s);
            atomicAdd(&chq[c], q);
        }
    }
}

// ---------------------------------------------------------------------------
// Kernel 6: BN finalize: mean / rsqrt(var + eps) per channel
// ---------------------------------------------------------------------------
__global__ void k_bnfin(const float* __restrict__ chs, const float* __restrict__ chq,
                        float* __restrict__ mean_a, float* __restrict__ inv_a) {
    int c = blockIdx.x * blockDim.x + threadIdx.x;
    if (c < C_) {
        float n = (float)(B_ * T_);
        float m = chs[c] / n;
        float v = chq[c] / n - m * m;
        mean_a[c] = m;
        inv_a[c]  = rsqrtf(v + EPS_);
    }
}

// ---------------------------------------------------------------------------
// Kernel 7: out = gamma*(wy - mean)*inv + beta + x_this
// ---------------------------------------------------------------------------
__global__ void k_apply(const float* __restrict__ wy, const float* __restrict__ xt,
                        const float* __restrict__ gamma, const float* __restrict__ beta,
                        const float* __restrict__ mean_a, const float* __restrict__ inv_a,
                        float* __restrict__ out) {
    size_t i = (size_t)blockIdx.x * blockDim.x + threadIdx.x;
    int c = (int)((i / T_) & (C_ - 1));
    out[i] = gamma[c] * (wy[i] - mean_a[c]) * inv_a[c] + beta[c] + xt[i];
}

// ---------------------------------------------------------------------------
extern "C" void kernel_launch(void* const* d_in, const int* in_sizes, int n_in,
                              void* d_out, int out_size, void* d_ws, size_t ws_size,
                              hipStream_t stream) {
    (void)in_sizes; (void)n_in; (void)out_size; (void)ws_size;
    const float* x_this  = (const float*)d_in[0];
    const float* x_other = (const float*)d_in[1];
    const float* Wg      = (const float*)d_in[2];
    const float* bg      = (const float*)d_in[3];
    const float* Wth     = (const float*)d_in[4];
    const float* bth     = (const float*)d_in[5];
    const float* Wph     = (const float*)d_in[6];
    const float* bph     = (const float*)d_in[7];
    const float* Wz      = (const float*)d_in[8];
    const float* bz      = (const float*)d_in[9];
    const float* gamma   = (const float*)d_in[10];
    const float* beta    = (const float*)d_in[11];
    float* out = (float*)d_out;

    // ---- workspace carve (all sub-buffers 256B aligned) ----
    char* w = (char*)d_ws;
    auto carve = [&](size_t bytes) {
        char* p = w;
        w += (bytes + 255) & ~(size_t)255;
        return p;
    };
    _Float16* xth    = (_Float16*)carve((size_t)B_ * T_ * C_ * 2);
    _Float16* xoh    = (_Float16*)carve((size_t)B_ * T_ * C_ * 2);
    _Float16* Wg_h   = (_Float16*)carve((size_t)CI_ * C_ * 2);
    _Float16* Wt_h   = (_Float16*)carve((size_t)CI_ * C_ * 2);
    _Float16* Wp_h   = (_Float16*)carve((size_t)CI_ * C_ * 2);
    _Float16* Wz_h   = (_Float16*)carve((size_t)C_ * CI_ * 2);
    _Float16* theta_h= (_Float16*)carve((size_t)B_ * T_ * CI_ * 2);
    _Float16* phi_h  = (_Float16*)carve((size_t)B_ * T_ * CI_ * 2);
    _Float16* g_h    = (_Float16*)carve((size_t)B_ * CI_ * T_ * 2);
    _Float16* y_h    = (_Float16*)carve((size_t)B_ * T_ * CI_ * 2);
    float*    wy     = (float*)   carve((size_t)B_ * C_ * T_ * 4);
    float*    stats  = (float*)   carve((size_t)4 * C_ * 4);
    float* chs    = stats;
    float* chq    = stats + C_;
    float* mean_a = stats + 2 * C_;
    float* inv_a  = stats + 3 * C_;

    // ---- pipeline ----
    k_zero <<<dim3((2 * C_ + 255) / 256), dim3(256), 0, stream>>>(chs, 2 * C_);
    k_castw<<<dim3((CI_ * C_) / 256), dim3(256), 0, stream>>>(
        Wg, Wth, Wph, Wz, Wg_h, Wt_h, Wp_h, Wz_h);
    k_castx<<<dim3(T_ / 32, C_ / 32, B_), dim3(32, 8), 0, stream>>>(
        x_this, x_other, xth, xoh);
    k_proj <<<dim3(B_ * (CI_ / 16) * (T_ / 16) / 4), dim3(128), 0, stream>>>(
        xth, xoh, Wg_h, Wt_h, Wp_h, bg, bth, bph, theta_h, phi_h, g_h);
    k_flash<<<dim3(B_ * (T_ / 16) / 4), dim3(128), 0, stream>>>(
        theta_h, phi_h, g_h, y_h);
    k_wz   <<<dim3(B_ * (C_ / 16) * (T_ / 16) / 4), dim3(128), 0, stream>>>(
        y_h, Wz_h, bz, wy, chs, chq);
    k_bnfin<<<dim3((C_ + 255) / 256), dim3(256), 0, stream>>>(
        chs, chq, mean_a, inv_a);
    k_apply<<<dim3((B_ * C_ * T_) / 256), dim3(256), 0, stream>>>(
        wy, x_this, gamma, beta, mean_a, inv_a, out);
}
